// SPConv_42468636622922
// MI455X (gfx1250) — matrix operand order repaired and checked by reading, not compiled, not run on hardware
//
#include <hip/hip_runtime.h>
#include <hip/hip_bf16.h>

// ---------------------------------------------------------------------------
// Sparse-conv U-Net for MI455X (gfx1250, wave32).
// gather -> f16 WMMA GEMM (f32 accum) -> atomic scatter-add.
//
// Weights are pre-swizzled into B-fragment order (2x global_load_b128/lane),
// activations live in f16 shadows so A fragments are direct b128 gathers.
// Inner loop: 6 vmem b128 loads + 2 v_wmma_f32_16x16x32_f16, no LDS, no
// barriers.  Block = 128 threads (4 waves) = 32 pairs x 64 output channels.
// ---------------------------------------------------------------------------

typedef __attribute__((ext_vector_type(16))) _Float16 v16h;
typedef __attribute__((ext_vector_type(8)))  float    v8f;

// ---------------------------------------------------------------------------
// Core gathered-GEMM kernel
// ---------------------------------------------------------------------------
__global__ __launch_bounds__(128)
void gconv_wmma_kernel(const _Float16* __restrict__ X,  // f16 activations, row stride = xstride
                       int xstride,
                       const uint4*   __restrict__ Wsw, // swizzled f16 weights, this offset
                       const int*     __restrict__ ii,  // gather rows (null => identity)
                       const int*     __restrict__ oo,  // scatter rows (null => identity)
                       int npairs, int kchunks, int Cout, int ntiles,
                       float* __restrict__ OUT)
{
    __shared__ int outRow[32];

    const int tid   = threadIdx.x;
    const int lane  = tid & 31;
    const int wave  = tid >> 5;                 // 0..3
    const int tileM = blockIdx.x * 32;
    const int ntile = blockIdx.y * 4 + wave;    // 16-col tile (uniform per wave)
    const bool wactive = (ntile < ntiles);      // wave-uniform => EXEC stays all-ones

    if (tid < 32) {
        int p = tileM + tid;
        outRow[tid] = (p < npairs) ? (oo ? oo[p] : p) : -1;
    }
    __syncthreads();

    // Per-lane gather rows (tail lanes clamp to a valid row; their results are
    // masked at scatter time via outRow < 0, so garbage contents are harmless).
    const int m  = lane & 15;
    int p0 = tileM + m;       if (p0 >= npairs) p0 = npairs - 1;
    int p1 = tileM + 16 + m;  if (p1 >= npairs) p1 = npairs - 1;
    const int r0 = ii ? ii[p0] : p0;
    const int r1 = ii ? ii[p1] : p1;

    // A-fragment addressing (16-bit A 16x32): lanes 0-15 -> K groups 0..7 &
    // 16..23; lanes 16-31 -> 8..15 & 24..31.  Both groups are contiguous f16
    // runs => one b128 load each.
    const int kb = (lane >> 4) << 3;            // 0 or 8
    const _Float16* a0p = X + (size_t)r0 * xstride + kb;
    const _Float16* a1p = X + (size_t)r1 * xstride + kb;
    // B fragments: pre-swizzled [ntile][kchunk][lane][8 dwords]
    const uint4* bp = Wsw + ((size_t)ntile * kchunks * 32 + lane) * 2;

    v8f acc0 = {};
    v8f acc1 = {};

    if (wactive) {
#pragma unroll 2
        for (int c = 0; c < kchunks; ++c) {
            union { v16h v; uint4 q[2]; } af0, af1, bf;
            af0.q[0] = *(const uint4*)(a0p);        // K = kb .. kb+7
            af0.q[1] = *(const uint4*)(a0p + 16);   // K = kb+16 .. kb+23
            af1.q[0] = *(const uint4*)(a1p);
            af1.q[1] = *(const uint4*)(a1p + 16);
            bf.q[0]  = bp[0];
            bf.q[1]  = bp[1];
            acc0 = __builtin_amdgcn_wmma_f32_16x16x32_f16(
                false, af0.v, false, bf.v, (short)0, acc0, false, false);
            acc1 = __builtin_amdgcn_wmma_f32_16x16x32_f16(
                false, af1.v, false, bf.v, (short)0, acc1, false, false);
            a0p += 32; a1p += 32;     // next K chunk
            bp  += 64;                // 32 lanes * 2 uint4
        }

        // scatter: C/D layout -> lane l, VGPR r: M = r + 8*(l>>4), N = l&15
        const int col = (ntile << 4) + m;
        if (col < Cout) {
            const int mb = (lane >> 4) << 3;
#pragma unroll
            for (int r = 0; r < 8; ++r) {
                int d0 = outRow[mb + r];
                if (d0 >= 0) atomicAdd(&OUT[(size_t)d0 * Cout + col], acc0[r]);
                int d1 = outRow[16 + mb + r];
                if (d1 >= 0) atomicAdd(&OUT[(size_t)d1 * Cout + col], acc1[r]);
            }
        }
    }
}

// ---------------------------------------------------------------------------
// Weight swizzle: (K,Cin,Cout) f32 -> per offset [ntile][kchunk][lane][8dw]
// f16 pairs, i.e. the exact B-fragment VGPR image (zero-padded).
// ---------------------------------------------------------------------------
__global__ void wswizzle_kernel(const float* __restrict__ W, unsigned* __restrict__ Wsw,
                                int K, int Cin, int Cout, int ntiles, int kchunks)
{
    size_t total = (size_t)K * ntiles * kchunks * 256;   // dwords
    size_t i = (size_t)blockIdx.x * blockDim.x + threadIdx.x;
    if (i >= total) return;
    int d    = (int)(i & 7);
    int lane = (int)((i >> 3) & 31);
    size_t rest = i >> 8;
    int c = (int)(rest % kchunks); rest /= kchunks;
    int t = (int)(rest % ntiles);
    int k = (int)(rest / ntiles);

    int n  = (t << 4) + (lane & 15);
    int kk = (c << 5) + ((lane >> 4) << 3) + ((d >> 2) << 4) + ((d & 3) << 1);
    const float* Wk = W + (size_t)k * Cin * Cout;
    float f0 = (kk     < Cin && n < Cout) ? Wk[(size_t)kk * Cout + n]       : 0.f;
    float f1 = (kk + 1 < Cin && n < Cout) ? Wk[(size_t)(kk + 1) * Cout + n] : 0.f;
    union { _Float16 h[2]; unsigned u; } pk;
    pk.h[0] = (_Float16)f0;
    pk.h[1] = (_Float16)f1;
    Wsw[i] = pk.u;
}

// ---------------------------------------------------------------------------
// Support kernels
// ---------------------------------------------------------------------------
__global__ void bias_init_kernel(float* __restrict__ Y, const float* __restrict__ b,
                                 int M, int C)
{
    size_t i = (size_t)blockIdx.x * blockDim.x + threadIdx.x;
    if (i < (size_t)M * C) Y[i] = b[i % C];
}

__global__ __launch_bounds__(256)
void bn_stats_kernel(const float* __restrict__ X, int M, int C,
                     float* __restrict__ stats)   // [0..C)=mean, [C..2C)=var
{
    __shared__ float sS[256];
    __shared__ float sQ[256];
    int c = blockIdx.x;
    float s = 0.f, q = 0.f;
    for (int r = threadIdx.x; r < M; r += 256) {
        float v = X[(size_t)r * C + c];
        s += v; q += v * v;
    }
    sS[threadIdx.x] = s; sQ[threadIdx.x] = q;
    __syncthreads();
    for (int off = 128; off > 0; off >>= 1) {
        if (threadIdx.x < off) {
            sS[threadIdx.x] += sS[threadIdx.x + off];
            sQ[threadIdx.x] += sQ[threadIdx.x + off];
        }
        __syncthreads();
    }
    if (threadIdx.x == 0) {
        float mn = sS[0] / (float)M;
        float v  = sQ[0] / (float)M - mn * mn;
        stats[c]     = mn;
        stats[C + c] = v > 0.f ? v : 0.f;
    }
}

// BN + PReLU in place; optionally emits an f16 shadow (row stride = C).
__global__ void bn_apply_kernel(float* __restrict__ X, const float* __restrict__ stats,
                                const float* __restrict__ g, const float* __restrict__ bt,
                                const float* __restrict__ a, int M, int C, int act,
                                _Float16* __restrict__ X16)
{
    size_t i = (size_t)blockIdx.x * blockDim.x + threadIdx.x;
    if (i >= (size_t)M * C) return;
    int c = (int)(i % C);
    float y = (X[i] - stats[c]) * rsqrtf(stats[C + c] + 1e-5f) * g[c] + bt[c];
    if (act) y = (y >= 0.f) ? y : a[c] * y;
    X[i] = y;
    if (X16) X16[i] = (_Float16)y;
}

// Concat two f32 buffers into an f16 buffer (stride Ca+Cb).
__global__ void concat16_kernel(const float* __restrict__ A, int Ca,
                                const float* __restrict__ Bp, int Cb,
                                _Float16* __restrict__ Y, int M)
{
    int C = Ca + Cb;
    size_t i = (size_t)blockIdx.x * blockDim.x + threadIdx.x;
    if (i >= (size_t)M * C) return;
    size_t r = i / C;
    int    c = (int)(i % C);
    float v = (c < Ca) ? A[r * Ca + c] : Bp[r * Cb + (c - Ca)];
    Y[i] = (_Float16)v;
}

// (B,3,N) f32 -> (B*N, 32) f16: [1, fx*2/180, fy*2/180, fz*2/60, 0...0]
__global__ void preprocess16_kernel(const float* __restrict__ F,
                                    _Float16* __restrict__ Y, int Bn, int N)
{
    size_t i = (size_t)blockIdx.x * blockDim.x + threadIdx.x;
    if (i >= (size_t)Bn * N * 32) return;
    int c = (int)(i & 31);
    size_t r = i >> 5;
    int b = (int)(r / N), n = (int)(r % N);
    float v = 0.f;
    if (c == 0) v = 1.f;
    else if (c < 4) {
        float s = (c == 3) ? (1.f / 30.f) : (1.f / 90.f);
        v = F[(size_t)b * 3 * N + (size_t)(c - 1) * N + n] * s;
    }
    Y[i] = (_Float16)v;
}

__global__ void transpose_out_kernel(const float* __restrict__ L,  // (B*N,C)
                                     float* __restrict__ O,        // (B,C,N)
                                     int Bn, int N, int C)
{
    size_t i = (size_t)blockIdx.x * blockDim.x + threadIdx.x;
    if (i >= (size_t)Bn * N * C) return;
    int c = (int)(i % C);
    size_t r = i / C;
    int b = (int)(r / N), n = (int)(r % N);
    O[(size_t)b * C * N + (size_t)c * N + n] = L[i];
}

// ---------------------------------------------------------------------------
// Host orchestration
// ---------------------------------------------------------------------------
extern "C" void kernel_launch(void* const* d_in, const int* in_sizes, int n_in,
                              void* d_out, int out_size, void* d_ws, size_t ws_size,
                              hipStream_t stream)
{
    (void)n_in; (void)out_size; (void)ws_size;
    int idx = 0;
    auto nextf = [&]() { return (const float*)d_in[idx++]; };

    // setup_inputs() insertion order: features, xyz, params{W,b,g,bt,a}x24, rules
    const float* features = nextf();
    const float* xyz = nextf(); (void)xyz;

    struct ConvP { const float *W, *b, *g, *bt, *a; };
    ConvP P[24];
    for (int i = 0; i < 24; ++i) {
        P[i].W = nextf(); P[i].b = nextf(); P[i].g = nextf();
        P[i].bt = nextf(); P[i].a = nextf();
    }
    enum { c0a, c0b, s0a, s0b, s1a, s1b, s1c, s2a, s2b, s2c, s3a, s3b, s3c, s3d,
           f3a, f3b, f2a, f2b, f1a, f1b, f0a, f0b, fca, fcb };

    struct Pr { const int* ii; const int* oo; int len; };
    auto read_pairs = [&](Pr* dst, int K) {
        for (int k = 0; k < K; ++k) {
            dst[k].len = in_sizes[idx];
            dst[k].ii  = (const int*)d_in[idx++];
            dst[k].oo  = (const int*)d_in[idx++];
        }
    };
    static Pr l1[27], l2[27], l3[27], l4[27], d00[8], d10[27], d20[27], d30[27],
              u00[8], u10[27], u20[27], u30[27];
    read_pairs(l1, 27); read_pairs(l2, 27); read_pairs(l3, 27); read_pairs(l4, 27);
    read_pairs(d00, 8); read_pairs(d10, 27); read_pairs(d20, 27); read_pairs(d30, 27);
    read_pairs(u00, 8); read_pairs(u10, 27); read_pairs(u20, 27); read_pairs(u30, 27);

    const int n0 = in_sizes[0] / 3;   // B*N (features is B,3,N)
    const int Bn = 2;
    const int N  = n0 / Bn;
    // submanifold center offset (k=13) is the identity map => length == #rows
    const int n1 = l1[13].len, n2 = l2[13].len, n3 = l3[13].len, n4 = l4[13].len;

    // ---- workspace bump allocator (bytes) ----
    char* wsb = (char*)d_ws; size_t off = 0;
    auto alloc = [&](size_t bytes) -> void* {
        void* p = (void*)(wsb + off);
        off += (bytes + 255) & ~(size_t)255;
        return p;
    };
    size_t gsf = (size_t)n0 * 64;
    auto mx = [&](size_t v) { if (v > gsf) gsf = v; };
    mx((size_t)n1 * 128); mx((size_t)n2 * 256); mx((size_t)n3 * 512); mx((size_t)n4 * 512);

    float*    S0f = (float*)alloc(gsf * 4);
    float*    S1f = (float*)alloc(gsf * 4);
    _Float16* T0h = (_Float16*)alloc(gsf * 2);
    _Float16* T1h = (_Float16*)alloc(gsf * 2);
    float*    L0f = (float*)alloc((size_t)n0 * 32 * 4);
    float*    L1f = (float*)alloc((size_t)n1 * 64 * 4);
    float*    L2f = (float*)alloc((size_t)n2 * 128 * 4);
    float*    L3f = (float*)alloc((size_t)n3 * 256 * 4);
    _Float16* L0h = (_Float16*)alloc((size_t)n0 * 32 * 2);
    _Float16* L1h = (_Float16*)alloc((size_t)n1 * 64 * 2);
    _Float16* L2h = (_Float16*)alloc((size_t)n2 * 128 * 2);
    _Float16* L3h = (_Float16*)alloc((size_t)n3 * 256 * 2);
    _Float16* F0h = (_Float16*)alloc((size_t)n0 * 32 * 2);
    float*    stats = (float*)alloc(1024 * 4);
    unsigned* Wsw = (unsigned*)alloc((size_t)27 * 32 * 16 * 256 * 4);  // max layer

    // ---- launch helpers ----
    auto conv = [&](const _Float16* X16, int xstride, const ConvP& p, const Pr* pr,
                    int K, int Mout, int Cin, int Cout, float* Y) {
        int ntiles  = (Cout + 15) / 16;
        int kchunks = (Cin + 31) / 32;
        size_t swdw = (size_t)K * ntiles * kchunks * 256;
        wswizzle_kernel<<<(unsigned)((swdw + 255) / 256), 256, 0, stream>>>(
            p.W, Wsw, K, Cin, Cout, ntiles, kchunks);
        size_t tot = (size_t)Mout * Cout;
        bias_init_kernel<<<(unsigned)((tot + 255) / 256), 256, 0, stream>>>(Y, p.b, Mout, Cout);
        size_t offsetStride = (size_t)ntiles * kchunks * 64;  // uint4 per offset
        for (int k = 0; k < K; ++k) {
            int np = pr ? pr[k].len : Mout;
            if (np <= 0) continue;
            dim3 g((unsigned)((np + 31) / 32), (unsigned)((Cout + 63) / 64));
            gconv_wmma_kernel<<<g, 128, 0, stream>>>(
                X16, xstride, (const uint4*)Wsw + (size_t)k * offsetStride,
                pr ? pr[k].ii : nullptr, pr ? pr[k].oo : nullptr,
                np, kchunks, Cout, ntiles, Y);
        }
    };
    auto bnact = [&](float* X, const ConvP& p, int M, int C, _Float16* X16) {
        bn_stats_kernel<<<(unsigned)C, 256, 0, stream>>>(X, M, C, stats);
        size_t tot = (size_t)M * C;
        bn_apply_kernel<<<(unsigned)((tot + 255) / 256), 256, 0, stream>>>(
            X, stats, p.g, p.bt, p.a, M, C, 1, X16);
    };
    auto cat = [&](const float* A, int Ca, const float* Bp, int Cb, _Float16* Y, int M) {
        size_t tot = (size_t)M * (Ca + Cb);
        concat16_kernel<<<(unsigned)((tot + 255) / 256), 256, 0, stream>>>(A, Ca, Bp, Cb, Y, M);
    };

    // ---- forward pass ----
    {
        size_t tot = (size_t)n0 * 32;
        preprocess16_kernel<<<(unsigned)((tot + 255) / 256), 256, 0, stream>>>(features, F0h, Bn, N);
    }

    conv(F0h, 32,  P[c0a], nullptr, 1, n0, 4, 32, S0f);    bnact(S0f, P[c0a], n0, 32, T0h);
    conv(T0h, 32,  P[c0b], nullptr, 1, n0, 32, 32, L0f);   bnact(L0f, P[c0b], n0, 32, L0h);
    conv(L0h, 32,  P[s0a], nullptr, 1, n0, 32, 32, S0f);   bnact(S0f, P[s0a], n0, 32, T0h);
    conv(T0h, 32,  P[s0b], d00, 8,  n1, 32, 64, L1f);      bnact(L1f, P[s0b], n1, 64, L1h);
    conv(L1h, 64,  P[s1a], l1, 27,  n1, 64, 64, S0f);      bnact(S0f, P[s1a], n1, 64, T0h);
    conv(T0h, 64,  P[s1b], l1, 27,  n1, 64, 128, S1f);     bnact(S1f, P[s1b], n1, 128, T1h);
    conv(T1h, 128, P[s1c], d10, 27, n2, 128, 128, L2f);    bnact(L2f, P[s1c], n2, 128, L2h);
    conv(L2h, 128, P[s2a], l2, 27,  n2, 128, 128, S0f);    bnact(S0f, P[s2a], n2, 128, T0h);
    conv(T0h, 128, P[s2b], l2, 27,  n2, 128, 256, S1f);    bnact(S1f, P[s2b], n2, 256, T1h);
    conv(T1h, 256, P[s2c], d20, 27, n3, 256, 256, L3f);    bnact(L3f, P[s2c], n3, 256, L3h);
    conv(L3h, 256, P[s3a], l3, 27,  n3, 256, 512, S0f);    bnact(S0f, P[s3a], n3, 512, T0h);
    conv(T0h, 512, P[s3b], d30, 27, n4, 512, 512, S1f);    bnact(S1f, P[s3b], n4, 512, T1h);
    conv(T1h, 512, P[s3c], l4, 27,  n4, 512, 512, S0f);    bnact(S0f, P[s3c], n4, 512, T0h);
    conv(T0h, 512, P[s3d], l4, 27,  n4, 512, 512, S1f);    bnact(S1f, P[s3d], n4, 512, T1h); // l4
    conv(T1h, 512, P[f3a], u30, 27, n3, 512, 512, S0f);    bnact(S0f, P[f3a], n3, 512, T0h);
    conv(T0h, 512, P[f3b], l3, 27,  n3, 512, 256, S1f);    bnact(S1f, P[f3b], n3, 256, nullptr); // u3
    cat(L3f, 256, S1f, 256, T0h, n3);
    conv(T0h, 512, P[f2a], u20, 27, n2, 512, 256, S0f);    bnact(S0f, P[f2a], n2, 256, T1h);
    conv(T1h, 256, P[f2b], l2, 27,  n2, 256, 128, S1f);    bnact(S1f, P[f2b], n2, 128, nullptr); // u2
    cat(L2f, 128, S1f, 128, T0h, n2);
    conv(T0h, 256, P[f1a], u10, 27, n1, 256, 128, S0f);    bnact(S0f, P[f1a], n1, 128, T1h);
    conv(T1h, 128, P[f1b], l1, 27,  n1, 128, 64, S1f);     bnact(S1f, P[f1b], n1, 64, nullptr);  // u1
    cat(L1f, 64, S1f, 64, T0h, n1);
    conv(T0h, 128, P[f0a], u00, 8,  n0, 128, 64, S0f);     bnact(S0f, P[f0a], n0, 64, T1h);
    conv(T1h, 64,  P[f0b], nullptr, 1, n0, 64, 32, S1f);   bnact(S1f, P[f0b], n0, 32, nullptr);  // u0
    cat(L0f, 32, S1f, 32, T0h, n0);
    conv(T0h, 64,  P[fca], nullptr, 1, n0, 64, 32, S0f);   bnact(S0f, P[fca], n0, 32, T1h);
    conv(T1h, 32,  P[fcb], nullptr, 1, n0, 32, 19, S1f);   // logits, no BN

    {
        size_t tot = (size_t)n0 * 19;
        transpose_out_kernel<<<(unsigned)((tot + 255) / 256), 256, 0, stream>>>(
            S1f, (float*)d_out, Bn, N, 19);
    }
}
